// TorchSpexModel_34333968564630
// MI455X (gfx1250) — compile-verified, matrix operand order checked by reading, not run on hardware
//
#include <hip/hip_runtime.h>

// ---------------- problem constants ----------------
#define NATOMS   10000
#define APAD     10016            // padded to multiple of 32
#define NSTRUCT  100
#define FDIM     4096             // (LMAX+1)*Q*Q
#define NHID     256
#define NPSEUDO  4
#define NTOT     1024             // NPSEUDO*NHID

// ---------------- CDNA5 types ----------------
typedef __attribute__((ext_vector_type(16))) __bf16        v16bf;
typedef __attribute__((ext_vector_type(8)))  float         v8f;
typedef __attribute__((ext_vector_type(4)))  unsigned int  v4u;
typedef __attribute__((ext_vector_type(8)))  int           v8i;
typedef __attribute__((ext_vector_type(4)))  int           v4i;

struct Frag32B { uint4 lo; uint4 hi; };

// fp32 -> bf16 (round to nearest even), bit-level
__device__ __forceinline__ unsigned short f2bf(float x) {
    unsigned int u = __float_as_uint(x);
    unsigned int r = (u + 0x7FFFu + ((u >> 16) & 1u)) >> 16;
    return (unsigned short)r;
}

// Load one 16x32 bf16 WMMA operand fragment from an LDS tile whose rows have a
// stride of 40 bf16 elements (80B) -- produced by TDM row padding (64B data +
// 16B pad). Per CDNA5 ISA 16-bit A/B layout:
//   lanes 0-15 : row = lane,    K = 0..7   and 16..23
//   lanes 16-31: row = lane-16, K = 8..15  and 24..31
__device__ __forceinline__ v16bf ld_frag(const unsigned short* base, int lane) {
    int hi = lane >> 4;
    int r  = lane & 15;
    const char* p = (const char*)(base + r * 40);
    Frag32B t;
    t.lo = *(const uint4*)(p + hi * 16);
    t.hi = *(const uint4*)(p + 32 + hi * 16);
    return __builtin_bit_cast(v16bf, t);
}

__device__ __forceinline__ float silu(float x) {
    return x / (1.0f + __expf(-x));
}

// ---------------- TDM: 2-D bf16 tile load, global -> LDS ----------------
// Builds a Tensor DMA Descriptor (D#) per CDNA5 ISA:
//   group0: count=1 | lds_addr | global_addr[56:0] | type=2
//   group1: data_size=2B, pad_enable, pad_interval=16 DW (64B), pad_amount=4 DW
//           (16B)  ->  LDS row stride = 80B = 40 bf16, matching ld_frag().
// All dims/strides in data_size (bf16) units. 2-D tile: groups 2/3 zero.
// 6-arg builtin form (amdgpu-toolchain clang-23).
__device__ __forceinline__ void tdm_load_2d(const void* gsrc, unsigned lds_off,
                                            unsigned tensor_d0, unsigned tensor_d1,
                                            unsigned tile_d0,   unsigned tile_d1,
                                            unsigned stride0) {
    unsigned long long ga = (unsigned long long)(uintptr_t)gsrc;
    v4u g0;
    g0[0] = 1u;                                                   // count=1, user
    g0[1] = lds_off;                                              // lds_addr (bytes)
    g0[2] = (unsigned)ga;                                         // global_addr lo
    g0[3] = (unsigned)((ga >> 32) & 0x01FFFFFFu) | (2u << 30);    // hi + type=2
    v8i g1;
    g1[0] = (int)((1u << 16)        // data_size: 2 bytes
                | (1u << 20)        // pad_enable
                | (3u << 22)        // pad_interval: 16 DWORDs between pads
                | (3u << 25));      // pad_amount:   4 DWORDs
    g1[1] = (int)((tensor_d0 & 0xFFFFu) << 16);                   // tdim0[15:0]
    g1[2] = (int)((tensor_d0 >> 16) | ((tensor_d1 & 0xFFFFu) << 16));
    g1[3] = (int)((tensor_d1 >> 16) | (tile_d0 << 16));           // tile_dim0
    g1[4] = (int)tile_d1;                                         // tile_dim1 (dim2=0)
    g1[5] = (int)stride0;                                         // dim0 stride lo32
    g1[6] = 0;
    g1[7] = 0;
    v4i z4 = {0, 0, 0, 0};
    v8i z8 = {0, 0, 0, 0, 0, 0, 0, 0};
    __builtin_amdgcn_tensor_load_to_lds(g0, g1, z4, z4, z8, 0);
}

// ---------------- kernel 0a: W0 [4,4096,256] f32 -> W0B bf16 [N=1024][K=4096] ----
__global__ void k_convW0(const float* __restrict__ W0, unsigned short* __restrict__ W0B) {
    long i = (long)blockIdx.x * 256 + threadIdx.x;       // over 1024*4096
    if (i >= (long)NTOT * FDIM) return;
    int n = (int)(i >> 12);          // 0..1023  (p*256 + h)
    int k = (int)(i & 4095);         // feature index f
    int p = n >> 8, h = n & 255;
    W0B[i] = f2bf(W0[((long)p * FDIM + k) * NHID + h]);
}

// ---------------- kernel 0b: W1 [4,256,256] f32 -> W1B bf16 [N=1024][K=256] -----
__global__ void k_convW1(const float* __restrict__ W1, unsigned short* __restrict__ W1B) {
    int i = blockIdx.x * 256 + threadIdx.x;              // over 1024*256
    if (i >= NTOT * NHID) return;
    int n = i >> 8;                  // p*256 + h
    int k = i & 255;
    int p = n >> 8, h = n & 255;
    W1B[i] = f2bf(W1[(p * NHID + k) * NHID + h]);
}

// ---------------- kernel 1: power-spectrum features -> bf16 [APAD][4096] --------
__global__ void k_features(const float* __restrict__ c0, const float* __restrict__ c1,
                           const float* __restrict__ c2, const float* __restrict__ c3,
                           unsigned short* __restrict__ featB) {
    __shared__ float tile[8][16][32];                    // 16KB
    const int lane = threadIdx.x & 31;
    const int w    = threadIdx.x >> 5;
    const long a   = (long)blockIdx.x * 8 + w;           // < APAD by grid construction
    const bool pad = (a >= NATOMS);

    float reg[16];
    {
        const float* s0 = c0 + a * 32;
        const float* s1 = c1 + a * 96;
        const float* s2 = c2 + a * 160;
        const float* s3 = c3 + a * 224;
        reg[0] = pad ? 0.f : s0[lane];
#pragma unroll
        for (int m = 0; m < 3; ++m) reg[1 + m] = pad ? 0.f : s1[m * 32 + lane];
#pragma unroll
        for (int m = 0; m < 5; ++m) reg[4 + m] = pad ? 0.f : s2[m * 32 + lane];
#pragma unroll
        for (int m = 0; m < 7; ++m) reg[9 + m] = pad ? 0.f : s3[m * 32 + lane];
#pragma unroll
        for (int r = 0; r < 16; ++r) tile[w][r][lane] = reg[r];
    }
    __syncthreads();

    unsigned short* dst = featB + a * FDIM;
    const float cgl[4] = {1.0f, 0.57735026919f, 0.44721359550f, 0.37796447301f};
    int base = 0;
#pragma unroll
    for (int l = 0; l <= 3; ++l) {
        const int nm = 2 * l + 1;
        for (int q = 0; q < 32; ++q) {
            float s = 0.f;
            for (int m = 0; m < nm; ++m)
                s += tile[w][base + m][q] * reg[base + m];    // broadcast * own column
            dst[l * 1024 + q * 32 + lane] = f2bf(cgl[l] * s);
        }
        base += nm;
    }
}

// ---------------- kernel 2: h0 = silu( pw .* (feat @ W0) ), bf16 out ------------
// Block tile 32(M) x 256(N); 8 waves as 2x4. TDM stages double-buffered A/B tiles.
__global__ __launch_bounds__(256) void k_gemm0(const unsigned short* __restrict__ featB,
                                               const unsigned short* __restrict__ W0B,
                                               const int*   __restrict__ species,
                                               const float* __restrict__ combW,
                                               unsigned short* __restrict__ h0B) {
    __shared__ unsigned short ldsA[2][32 * 40];          //  5 KB
    __shared__ unsigned short ldsB[2][256 * 40];         // 40 KB
    __shared__ float ldsScale[32 * 4];

    const int t    = threadIdx.x;
    const int lane = t & 31, w = t >> 5;
    const int wm   = w >> 2, wn = w & 3;
    const long aBase = (long)blockIdx.x * 32;
    const int  n0    = blockIdx.y * 256;

    if (t < 32) {
        long a = aBase + t;
        int sp = (a < NATOMS) ? species[a] : -1;
#pragma unroll
        for (int p = 0; p < 4; ++p)
            ldsScale[t * 4 + p] = (sp >= 0) ? combW[p * 4 + sp] : 0.f;
    }

    const unsigned short* Abase = featB + aBase * FDIM;
    const unsigned short* Bbase = W0B + (long)n0 * FDIM;
    constexpr int NK = FDIM / 32;                        // 128 K-steps

    if (w == 0) {                                        // TDM prologue: steps 0,1
        tdm_load_2d(Abase +  0, (unsigned)(uintptr_t)&ldsA[0][0], FDIM,  32, 32,  32, FDIM);
        tdm_load_2d(Bbase +  0, (unsigned)(uintptr_t)&ldsB[0][0], FDIM, 256, 32, 256, FDIM);
        tdm_load_2d(Abase + 32, (unsigned)(uintptr_t)&ldsA[1][0], FDIM,  32, 32,  32, FDIM);
        tdm_load_2d(Bbase + 32, (unsigned)(uintptr_t)&ldsB[1][0], FDIM, 256, 32, 256, FDIM);
    }

    v8f acc[4] = {};
    int cur = 0;
    for (int kt = 0; kt < NK; ++kt) {
        if (w == 0) {
            if (kt + 1 < NK) __builtin_amdgcn_s_wait_tensorcnt(2);
            else             __builtin_amdgcn_s_wait_tensorcnt(0);
        }
        __syncthreads();                                 // tiles[cur] ready for all waves

        v16bf af = ld_frag(&ldsA[cur][wm * 16 * 40], lane);
#pragma unroll
        for (int j = 0; j < 4; ++j) {
            v16bf bf = ld_frag(&ldsB[cur][(wn * 64 + j * 16) * 40], lane);
            acc[j] = __builtin_amdgcn_wmma_f32_16x16x32_bf16(
                         false, af, false, bf, (short)0, acc[j], false, false);
        }
        __syncthreads();                                 // all waves done with tiles[cur]

        if (w == 0 && kt + 2 < NK) {                     // prefetch step kt+2 into [cur]
            int kk = (kt + 2) * 32;
            tdm_load_2d(Abase + kk, (unsigned)(uintptr_t)&ldsA[cur][0], FDIM,  32, 32,  32, FDIM);
            tdm_load_2d(Bbase + kk, (unsigned)(uintptr_t)&ldsB[cur][0], FDIM, 256, 32, 256, FDIM);
        }
        cur ^= 1;
    }

    // epilogue: scale by pw, SiLU, store bf16
    const int hi = lane >> 4, lp = lane & 15;
#pragma unroll
    for (int j = 0; j < 4; ++j) {
        const int n = n0 + wn * 64 + j * 16 + lp;
        const int p = n >> 8;
#pragma unroll
        for (int v = 0; v < 8; ++v) {
            const int  m = wm * 16 + v + hi * 8;
            const long a = aBase + m;
            float x = acc[j][v] * ldsScale[m * 4 + p];
            h0B[a * NTOT + n] = f2bf(silu(x));
        }
    }
}

// ---------------- kernel 3: h1 = silu(h0 @ W1[p]); E_a += h1 @ W2[p]; scatter ---
__global__ __launch_bounds__(256) void k_gemm1(const unsigned short* __restrict__ h0B,
                                               const unsigned short* __restrict__ W1B,
                                               const float* __restrict__ W2,
                                               const int*   __restrict__ structIds,
                                               float* __restrict__ out) {
    __shared__ unsigned short ldsA[2][32 * 40];
    __shared__ unsigned short ldsB[2][256 * 40];
    __shared__ float ldsRed[32];

    const int t    = threadIdx.x;
    const int lane = t & 31, w = t >> 5;
    const int wm   = w >> 2, wn = w & 3;
    const long aBase = (long)blockIdx.x * 32;
    const int  p     = blockIdx.y;           // pseudo-species = N-block
    const int  n0    = p * 256;

    if (t < 32) ldsRed[t] = 0.f;

    const unsigned short* Abase = h0B + aBase * NTOT + n0;   // K-slice of this p
    const unsigned short* Bbase = W1B + (long)n0 * NHID;
    constexpr int NK = NHID / 32;                            // 8 K-steps

    if (w == 0) {
        tdm_load_2d(Abase +  0, (unsigned)(uintptr_t)&ldsA[0][0], NHID,  32, 32,  32, NTOT);
        tdm_load_2d(Bbase +  0, (unsigned)(uintptr_t)&ldsB[0][0], NHID, 256, 32, 256, NHID);
        tdm_load_2d(Abase + 32, (unsigned)(uintptr_t)&ldsA[1][0], NHID,  32, 32,  32, NTOT);
        tdm_load_2d(Bbase + 32, (unsigned)(uintptr_t)&ldsB[1][0], NHID, 256, 32, 256, NHID);
    }

    v8f acc[4] = {};
    int cur = 0;
    for (int kt = 0; kt < NK; ++kt) {
        if (w == 0) {
            if (kt + 1 < NK) __builtin_amdgcn_s_wait_tensorcnt(2);
            else             __builtin_amdgcn_s_wait_tensorcnt(0);
        }
        __syncthreads();

        v16bf af = ld_frag(&ldsA[cur][wm * 16 * 40], lane);
#pragma unroll
        for (int j = 0; j < 4; ++j) {
            v16bf bf = ld_frag(&ldsB[cur][(wn * 64 + j * 16) * 40], lane);
            acc[j] = __builtin_amdgcn_wmma_f32_16x16x32_bf16(
                         false, af, false, bf, (short)0, acc[j], false, false);
        }
        __syncthreads();

        if (w == 0 && kt + 2 < NK) {
            int kk = (kt + 2) * 32;
            tdm_load_2d(Abase + kk, (unsigned)(uintptr_t)&ldsA[cur][0], NHID,  32, 32,  32, NTOT);
            tdm_load_2d(Bbase + kk, (unsigned)(uintptr_t)&ldsB[cur][0], NHID, 256, 32, 256, NHID);
        }
        cur ^= 1;
    }

    // fused layer 3: per-lane partial dot with W2, then cross-lane + cross-wave reduce
    const int hi = lane >> 4, lp = lane & 15;
    float part[8] = {0.f, 0.f, 0.f, 0.f, 0.f, 0.f, 0.f, 0.f};
#pragma unroll
    for (int j = 0; j < 4; ++j) {
        const int hcol = wn * 64 + j * 16 + lp;          // hidden index within p
        const float w2v = W2[p * NHID + hcol];
#pragma unroll
        for (int v = 0; v < 8; ++v)
            part[v] += silu(acc[j][v]) * w2v;
    }
#pragma unroll
    for (int v = 0; v < 8; ++v) {
        part[v] += __shfl_xor(part[v], 1);
        part[v] += __shfl_xor(part[v], 2);
        part[v] += __shfl_xor(part[v], 4);
        part[v] += __shfl_xor(part[v], 8);               // full within 16-lane half
    }
    if (lp == 0) {
#pragma unroll
        for (int v = 0; v < 8; ++v)
            atomicAdd(&ldsRed[wm * 16 + v + hi * 8], part[v]);
    }
    __syncthreads();

    if (t < 32) {
        long a = aBase + t;
        if (a < NATOMS)
            atomicAdd(&out[structIds[a]], ldsRed[t]);
    }
}

// ---------------- host launcher ----------------
extern "C" void kernel_launch(void* const* d_in, const int* in_sizes, int n_in,
                              void* d_out, int out_size, void* d_ws, size_t ws_size,
                              hipStream_t stream) {
    (void)in_sizes; (void)n_in; (void)ws_size;

    const float* c0      = (const float*)d_in[0];
    const float* c1      = (const float*)d_in[1];
    const float* c2      = (const float*)d_in[2];
    const float* c3      = (const float*)d_in[3];
    const int*   species = (const int*)  d_in[4];
    const int*   structs = (const int*)  d_in[5];
    const float* combW   = (const float*)d_in[6];
    const float* W0      = (const float*)d_in[7];
    const float* W1      = (const float*)d_in[8];
    const float* W2      = (const float*)d_in[9];
    float* out = (float*)d_out;

    // workspace partition (bf16 buffers), ~112 MB total
    unsigned short* featB = (unsigned short*)d_ws;                 // [APAD][4096]
    unsigned short* W0B   = featB + (size_t)APAD * FDIM;           // [1024][4096]
    unsigned short* W1B   = W0B   + (size_t)NTOT * FDIM;           // [1024][256]
    unsigned short* h0B   = W1B   + (size_t)NTOT * NHID;           // [APAD][1024]

    (void)hipMemsetAsync(d_out, 0, (size_t)out_size * sizeof(float), stream);

    k_convW0<<<(int)(((long)NTOT * FDIM + 255) / 256), 256, 0, stream>>>(W0, W0B);
    k_convW1<<<(NTOT * NHID + 255) / 256, 256, 0, stream>>>(W1, W1B);
    k_features<<<APAD / 8, 256, 0, stream>>>(c0, c1, c2, c3, featB);
    k_gemm0<<<dim3(APAD / 32, NTOT / 256), 256, 0, stream>>>(featB, W0B, species, combW, h0B);
    k_gemm1<<<dim3(APAD / 32, NPSEUDO), 256, 0, stream>>>(h0B, W1B, W2, structs, out);
}
